// S5MaskedEncoder_81406810129049
// MI455X (gfx1250) — matrix operand order repaired and checked by reading, not compiled, not run on hardware
//
#include <hip/hip_runtime.h>
#include <hip/hip_bf16.h>

// Problem constants (match reference)
#define BB    16
#define TT    4096
#define CC    128
#define DD    512     // TOKEN_DIM
#define HH    512     // HIDDEN
#define TTOK  1024
#define MT    32      // token rows per block tile
#define LDA   520     // padded bf16 LDS row stride (512 + 8)
#define NWAVE 8

#define NWB  (8 * DD * DD)        // W_bank elems (2,097,152)
#define NWE  (DD * HH)            // W_emb elems  (262,144)
#define NTOK (BB * TTOK * DD)     // token elems  (8,388,608)

typedef __attribute__((ext_vector_type(16))) __bf16       v16bf;
typedef __attribute__((ext_vector_type(8)))  float        v8f;
typedef __attribute__((ext_vector_type(4)))  unsigned int u32x4;
typedef __attribute__((ext_vector_type(8)))  int          i32x8;
typedef __attribute__((ext_vector_type(4)))  int          i32x4;

union FragU { v16bf v; uint4 q[2]; };

static __device__ __forceinline__ unsigned int pk_bf2(float a, float b) {
  unsigned short lo = __builtin_bit_cast(unsigned short, (__bf16)a);
  unsigned short hi = __builtin_bit_cast(unsigned short, (__bf16)b);
  return (unsigned int)lo | ((unsigned int)hi << 16);
}

// A fragment (16x32 bf16): lane lo = row; elems 0..15 hold
// k = hi*16 .. hi*16+15 (contiguous, matches B)  -> two b128 loads.
static __device__ __forceinline__ v16bf load_a(const __bf16* sA, int m, int k0, int hi) {
  FragU f;
  const __bf16* p = sA + m * LDA + k0 + hi * 16;
  f.q[0] = *(const uint4*)(p);        // k .. k+7
  f.q[1] = *(const uint4*)(p + 8);    // k+8 .. k+15
  return f.v;
}

// B fragment (32x16 bf16, K x N): lane lo = column n; lanes 0-15 hold
// K = 0..15, lanes 16-31 hold K = 16..31 (elems in K order) -> 16
// consecutive bf16 from row-major bf16 weights = two b128 loads.
static __device__ __forceinline__ v16bf load_b(const __bf16* __restrict__ W, int n, int k0, int hi) {
  FragU f;
  const __bf16* p = W + ((size_t)n << 9) + k0 + hi * 16;
  f.q[0] = *(const uint4*)(p);        // k .. k+7
  f.q[1] = *(const uint4*)(p + 8);    // k+8 .. k+15
  return f.v;
}

// ---- One-shot fp32 -> bf16 conversion of weights + masked tokens into ws ----
__global__ __launch_bounds__(256)
void prep_bf16(const float* __restrict__ Wb, const float* __restrict__ We,
               const float* __restrict__ x,  const int* __restrict__ lengths,
               __bf16* __restrict__ wsWb, __bf16* __restrict__ wsWe,
               __bf16* __restrict__ wsTok) {
  const int QWB = NWB / 4, QWE = NWE / 4, QT = NTOK / 4;
  int q = blockIdx.x * 256 + threadIdx.x;
  if (q >= QWB + QWE + QT) return;
  float4 v;
  __bf16* dst;
  if (q < QWB) {
    size_t e = (size_t)q * 4;
    v = *(const float4*)(Wb + e);
    dst = wsWb + e;
  } else if (q < QWB + QWE) {
    size_t e = (size_t)(q - QWB) * 4;
    v = *(const float4*)(We + e);
    dst = wsWe + e;
  } else {
    size_t e = (size_t)(q - QWB - QWE) * 4;
    int t  = (int)(e >> 9);            // flat token index
    int b  = t >> 10;
    int tr = t & (TTOK - 1);
    v = make_float4(0.f, 0.f, 0.f, 0.f);
    if (tr < lengths[b] / 4) v = *(const float4*)(x + e);   // zero-pad invalid rows
    dst = wsTok + e;
  }
  uint2 pk;
  pk.x = pk_bf2(v.x, v.y);
  pk.y = pk_bf2(v.z, v.w);
  *(uint2*)dst = pk;
}

__global__ __launch_bounds__(256)
void s5_fused_encoder(const __bf16* __restrict__ wsWb,
                      const __bf16* __restrict__ wsWe,
                      const __bf16* __restrict__ wsTok,
                      const int*   __restrict__ session_idx,
                      const float* __restrict__ b_bank,
                      const float* __restrict__ ln1_g,
                      const float* __restrict__ ln1_b,
                      const float* __restrict__ b_emb,
                      const float* __restrict__ ln2_g,
                      const float* __restrict__ ln2_b,
                      float* __restrict__ out) {
  __shared__ __bf16 sA[MT * LDA];        // token tile / h tile (bf16)
  __shared__ float2 sRed[NWAVE * MT];    // per-wave (sum, sumsq) partials
  __shared__ float2 sStats[MT];          // per-row (mean, rsigma)

  const int tid = threadIdx.x;
  const int w  = tid >> 5;
  const int l  = tid & 31;
  const int lo = l & 15;
  const int hi = l >> 4;

  const int blk  = blockIdx.x;
  const int bIdx = blk >> 5;             // 32 tiles per batch
  const int tile = blk & 31;
  const int t0   = tile * MT;

  const int sid = session_idx[bIdx];
  const __bf16* W1 = wsWb + (size_t)sid * DD * DD;
  const float*  b1 = b_bank + (size_t)sid * DD;

  // ---- Phase 0: TDM async tensor load of the 32x512 bf16 token tile into
  // LDS, with hardware row padding (pad_interval=256 DW, pad_amount=4 DW)
  // reproducing the LDA=520 bank-rotation layout. Wave 0 issues; TENSORcnt
  // is per-wave, so wave 0 waits then the block barrier releases everyone.
  if (w == 0) {
    unsigned long long lds64 = (unsigned long long)(uintptr_t)(void*)&sA[0];
    unsigned lds_addr = (unsigned)lds64;                      // LDS byte offset
    unsigned long long ga =
        (unsigned long long)(uintptr_t)(wsTok + ((size_t)(bIdx * TTOK + t0) << 9));
    u32x4 g0;
    g0[0] = 1u;                                               // count=1, user D#
    g0[1] = lds_addr;                                         // lds_addr
    g0[2] = (unsigned)(ga & 0xffffffffu);                     // global_addr lo
    g0[3] = (unsigned)((ga >> 32) & 0x01ffffffu) | (2u << 30);// addr hi | type=2
    i32x8 g1;
    g1[0] = (int)((1u << 16)      // data_size = 1 -> 2 bytes
                | (1u << 20)      // pad_enable
                | (7u << 22)      // pad_interval: 7 -> every 256 DWORDs (one row)
                | (3u << 25));    // pad_amount: 3 -> 4 DWORDs (8 bf16)
    g1[1] = (int)(512u << 16);    // tensor_dim0 = 512 (bits 79:48)
    g1[2] = (int)(32u << 16);     // tensor_dim1 = 32  (bits 111:80)
    g1[3] = (int)(512u << 16);    // tile_dim0   = 512 (bits 127:112)
    g1[4] = (int)(32u);           // tile_dim1 = 32, tile_dim2 = 0
    g1[5] = (int)(512u);          // tensor_dim0_stride = 512 (lo 32)
    g1[6] = 0;                    // stride0 hi / tensor_dim1_stride lo
    g1[7] = 0;
    i32x4 z4 = {0, 0, 0, 0};      // groups 2/3 unused (2D tile)
    i32x8 z8 = {0, 0, 0, 0, 0, 0, 0, 0};
    __builtin_amdgcn_tensor_load_to_lds(g0, g1, z4, z4, z8, 0);
    __builtin_amdgcn_s_wait_tensorcnt(0);
  }
  __syncthreads();

  const int n0 = w * 64;                 // this wave's 64-column slab
  v8f acc[2][4];
  #pragma unroll
  for (int mi = 0; mi < 2; ++mi)
    #pragma unroll
    for (int ni = 0; ni < 4; ++ni) acc[mi][ni] = (v8f){};

  // ---- GEMM1: aligned = tokens @ W1^T (A in LDS, B = bf16 weights in ws) ----
  for (int k0 = 0; k0 < DD; k0 += 32) {
    if (k0 + 32 < DD)
      __builtin_prefetch(W1 + (((size_t)(n0 + lo)) << 9) + k0 + 32, 0, 1);
    v16bf afr0 = load_a(sA, 0 * 16 + lo, k0, hi);
    v16bf afr1 = load_a(sA, 1 * 16 + lo, k0, hi);
    #pragma unroll
    for (int ni = 0; ni < 4; ++ni) {
      v16bf bfr = load_b(W1, n0 + ni * 16 + lo, k0, hi);
      acc[0][ni] = __builtin_amdgcn_wmma_f32_16x16x32_bf16(
          false, afr0, false, bfr, (short)0, acc[0][ni], false, false);
      acc[1][ni] = __builtin_amdgcn_wmma_f32_16x16x32_bf16(
          false, afr1, false, bfr, (short)0, acc[1][ni], false, false);
    }
  }

  // ---- bias + LayerNorm 1 (row stats across all 8 waves) ----
  #pragma unroll
  for (int ni = 0; ni < 4; ++ni) {
    float bia = b1[n0 + ni * 16 + lo];
    #pragma unroll
    for (int mi = 0; mi < 2; ++mi)
      #pragma unroll
      for (int r = 0; r < 8; ++r) acc[mi][ni][r] += bia;
  }
  #pragma unroll
  for (int mi = 0; mi < 2; ++mi)
    #pragma unroll
    for (int r = 0; r < 8; ++r) {
      float s = 0.f, ss = 0.f;
      #pragma unroll
      for (int ni = 0; ni < 4; ++ni) {
        float v = acc[mi][ni][r];
        s += v; ss += v * v;
      }
      #pragma unroll
      for (int msk = 1; msk < 16; msk <<= 1) {   // reduce within 16-lane half
        s  += __shfl_xor(s,  msk, 32);
        ss += __shfl_xor(ss, msk, 32);
      }
      if (lo == 0) sRed[w * MT + (mi * 16 + hi * 8 + r)] = make_float2(s, ss);
    }
  __syncthreads();
  if (tid < MT) {
    float s = 0.f, ss = 0.f;
    #pragma unroll
    for (int i = 0; i < NWAVE; ++i) {
      float2 p = sRed[i * MT + tid];
      s += p.x; ss += p.y;
    }
    float mean = s * (1.f / DD);
    float var  = ss * (1.f / DD) - mean * mean;
    sStats[tid] = make_float2(mean, rsqrtf(var + 1e-5f));
  }
  __syncthreads();

  // normalize, write h back to sA as bf16 (A matrix for GEMM2)
  #pragma unroll
  for (int ni = 0; ni < 4; ++ni) {
    int n = n0 + ni * 16 + lo;
    float gg = ln1_g[n], bbias = ln1_b[n];
    #pragma unroll
    for (int mi = 0; mi < 2; ++mi)
      #pragma unroll
      for (int r = 0; r < 8; ++r) {
        int m = mi * 16 + hi * 8 + r;
        float2 st = sStats[m];
        float v = (acc[mi][ni][r] - st.x) * st.y * gg + bbias;
        sA[m * LDA + n] = (__bf16)v;
      }
  }
  __syncthreads();

  // ---- GEMM2: h2 = h @ W_emb^T ----
  #pragma unroll
  for (int mi = 0; mi < 2; ++mi)
    #pragma unroll
    for (int ni = 0; ni < 4; ++ni) acc[mi][ni] = (v8f){};

  for (int k0 = 0; k0 < HH; k0 += 32) {
    if (k0 + 32 < HH)
      __builtin_prefetch(wsWe + (((size_t)(n0 + lo)) << 9) + k0 + 32, 0, 1);
    v16bf afr0 = load_a(sA, 0 * 16 + lo, k0, hi);
    v16bf afr1 = load_a(sA, 1 * 16 + lo, k0, hi);
    #pragma unroll
    for (int ni = 0; ni < 4; ++ni) {
      v16bf bfr = load_b(wsWe, n0 + ni * 16 + lo, k0, hi);
      acc[0][ni] = __builtin_amdgcn_wmma_f32_16x16x32_bf16(
          false, afr0, false, bfr, (short)0, acc[0][ni], false, false);
      acc[1][ni] = __builtin_amdgcn_wmma_f32_16x16x32_bf16(
          false, afr1, false, bfr, (short)0, acc[1][ni], false, false);
    }
  }

  // ---- bias + LayerNorm 2 ----
  #pragma unroll
  for (int ni = 0; ni < 4; ++ni) {
    float bia = b_emb[n0 + ni * 16 + lo];
    #pragma unroll
    for (int mi = 0; mi < 2; ++mi)
      #pragma unroll
      for (int r = 0; r < 8; ++r) acc[mi][ni][r] += bia;
  }
  __syncthreads();   // sRed reuse safe: all waves are past prior reads
  #pragma unroll
  for (int mi = 0; mi < 2; ++mi)
    #pragma unroll
    for (int r = 0; r < 8; ++r) {
      float s = 0.f, ss = 0.f;
      #pragma unroll
      for (int ni = 0; ni < 4; ++ni) {
        float v = acc[mi][ni][r];
        s += v; ss += v * v;
      }
      #pragma unroll
      for (int msk = 1; msk < 16; msk <<= 1) {
        s  += __shfl_xor(s,  msk, 32);
        ss += __shfl_xor(ss, msk, 32);
      }
      if (lo == 0) sRed[w * MT + (mi * 16 + hi * 8 + r)] = make_float2(s, ss);
    }
  __syncthreads();
  if (tid < MT) {
    float s = 0.f, ss = 0.f;
    #pragma unroll
    for (int i = 0; i < NWAVE; ++i) {
      float2 p = sRed[i * MT + tid];
      s += p.x; ss += p.y;
    }
    float mean = s * (1.f / HH);
    float var  = ss * (1.f / HH) - mean * mean;
    sStats[tid] = make_float2(mean, rsqrtf(var + 1e-5f));
  }
  __syncthreads();

  // normalize + store fp32 output
  float* obase = out + ((size_t)bIdx * TTOK + t0) * HH;
  #pragma unroll
  for (int ni = 0; ni < 4; ++ni) {
    int n = n0 + ni * 16 + lo;
    float gg = ln2_g[n], bbias = ln2_b[n];
    #pragma unroll
    for (int mi = 0; mi < 2; ++mi)
      #pragma unroll
      for (int r = 0; r < 8; ++r) {
        int m = mi * 16 + hi * 8 + r;
        float2 st = sStats[m];
        obase[(size_t)m * HH + n] = (acc[mi][ni][r] - st.x) * st.y * gg + bbias;
      }
  }
}

extern "C" void kernel_launch(void* const* d_in, const int* in_sizes, int n_in,
                              void* d_out, int out_size, void* d_ws, size_t ws_size,
                              hipStream_t stream) {
  (void)in_sizes; (void)n_in; (void)ws_size; (void)out_size;
  const float* x           = (const float*)d_in[0];
  const int*   lengths     = (const int*)d_in[1];
  const int*   session_idx = (const int*)d_in[2];
  const float* W_bank      = (const float*)d_in[3];
  const float* b_bank      = (const float*)d_in[4];
  const float* ln1_g       = (const float*)d_in[5];
  const float* ln1_b       = (const float*)d_in[6];
  const float* W_emb       = (const float*)d_in[7];
  const float* b_emb       = (const float*)d_in[8];
  const float* ln2_g       = (const float*)d_in[9];
  const float* ln2_b       = (const float*)d_in[10];
  // d_in[11] = raw_mask_token (unused: token_mask is all-False in reference)
  float* out = (float*)d_out;

  // workspace layout (bf16): [W_bank 4MB][W_emb 0.5MB][tokens 16MB]
  __bf16* wsWb  = (__bf16*)d_ws;
  __bf16* wsWe  = wsWb + (size_t)NWB;
  __bf16* wsTok = wsWe + (size_t)NWE;

  int quads = (NWB + NWE + NTOK) / 4;
  hipLaunchKernelGGL(prep_bf16, dim3((quads + 255) / 256), dim3(256), 0, stream,
                     W_bank, W_emb, x, lengths, wsWb, wsWe, wsTok);

  dim3 grid(BB * (TTOK / MT));   // 16 batches * 32 tiles = 512 blocks
  dim3 block(256);               // 8 waves (wave32)
  hipLaunchKernelGGL(s5_fused_encoder, grid, block, 0, stream,
                     wsWb, wsWe, wsTok, session_idx, b_bank,
                     ln1_g, ln1_b, b_emb, ln2_g, ln2_b, out);
}
